// MultiheadAttention_30966714204297
// MI455X (gfx1250) — matrix operand order repaired and checked by reading, not compiled, not run on hardware
//
#include <hip/hip_runtime.h>

// ---------------------------------------------------------------------------
// MI455X (gfx1250) multi-head attention, bf16 WMMA path, wave32.
// Compute-bound (~206 GFLOP vs ~100MB HBM traffic) -> all matmuls on
// v_wmma_f32_16x16x32_bf16.  GEMMs: 32x64 tile per wave, two-phase
// software-pipelined k-loop (no register rotation copies), launch_bounds to
// avoid spills.  Attention stages K tiles into LDS with
// global_load_async_to_lds_b128 (double buffered, s_wait_asynccnt).
// One wave per workgroup so EXEC is all-1s for every WMMA.
// ---------------------------------------------------------------------------

typedef __attribute__((ext_vector_type(16))) __bf16        v16bf;
typedef __attribute__((ext_vector_type(8)))  float         v8f;
typedef __attribute__((ext_vector_type(4)))  unsigned int  u32x4;

constexpr int B_ = 4, L_ = 2048, D_ = 1024, H_ = 16, HD_ = 64;

#define DEVINL __device__ __forceinline__

DEVINL unsigned short f2bf(float f) {
  unsigned int u = __builtin_bit_cast(unsigned int, f);
  u += 0x7fffu + ((u >> 16) & 1u);          // round-to-nearest-even
  return (unsigned short)(u >> 16);
}

DEVINL v8f vzero() { v8f z = {0.f,0.f,0.f,0.f,0.f,0.f,0.f,0.f}; return z; }

union OpU { v16bf v; u32x4 q[2]; };

// A-operand (16x32 bf16, ISA 7.12.2): lane = M + 16*(Kblock), per lane two
// contiguous 16B chunks of a row-major [rows][ld] bf16 matrix.
DEVINL v16bf load_A_g(const unsigned short* base, int ld, int row0, int col0) {
  int lane = threadIdx.x & 31;
  int m  = lane & 15;
  int kb = (lane >> 4) << 3;                // 0 or 8 halfwords
  const unsigned short* p = base + (size_t)(row0 + m) * ld + col0 + kb;
  OpU r;
  r.q[0] = *(const u32x4*)(p);              // K = kb .. kb+7
  r.q[1] = *(const u32x4*)(p + 16);         // K = kb+16 .. kb+23
  return r.v;
}

// B-operand (32x16 bf16): lane = N + 16*(Kblock); source is "Bt" row-major
// [n][k] (i.e. B transposed), two 8-halfword chunks per lane (K = kb..kb+7,
// kb+16..kb+23), matching the A layout.  Works for
// global or LDS-backed pointers (addrspace inferred after inlining).
DEVINL v16bf load_B_g(const unsigned short* base, int ld, int n0, int k0) {
  int lane = threadIdx.x & 31;
  int n  = lane & 15;
  int kb = (lane >> 4) << 3;                // 0 or 8 halfwords
  const unsigned short* p = base + (size_t)(n0 + n) * ld + k0 + kb;
  OpU r;
  r.q[0] = *(const u32x4*)(p);
  r.q[1] = *(const u32x4*)(p + 16);
  return r.v;
}

DEVINL v8f wmma_bf16(v16bf a, v16bf b, v8f c) {
  return __builtin_amdgcn_wmma_f32_16x16x32_bf16(false, a, false, b,
                                                 (short)0, c, false, false);
}

// Async-copy 128 bytes (one bf16 row of 64) global -> LDS, 8x b128.
// Same INST_OFFSET applies to both LDS dest and global src (ISA 08 §4.4).
DEVINL void async_copy_row128(const unsigned short* gsrc, unsigned short* ldst) {
  unsigned       lds = (unsigned)(size_t)ldst;          // low 32b = LDS offset
  unsigned long long ga = (unsigned long long)(size_t)gsrc;
  asm volatile("global_load_async_to_lds_b128 %0, %1, off"            :: "v"(lds), "v"(ga) : "memory");
  asm volatile("global_load_async_to_lds_b128 %0, %1, off offset:16"  :: "v"(lds), "v"(ga) : "memory");
  asm volatile("global_load_async_to_lds_b128 %0, %1, off offset:32"  :: "v"(lds), "v"(ga) : "memory");
  asm volatile("global_load_async_to_lds_b128 %0, %1, off offset:48"  :: "v"(lds), "v"(ga) : "memory");
  asm volatile("global_load_async_to_lds_b128 %0, %1, off offset:64"  :: "v"(lds), "v"(ga) : "memory");
  asm volatile("global_load_async_to_lds_b128 %0, %1, off offset:80"  :: "v"(lds), "v"(ga) : "memory");
  asm volatile("global_load_async_to_lds_b128 %0, %1, off offset:96"  :: "v"(lds), "v"(ga) : "memory");
  asm volatile("global_load_async_to_lds_b128 %0, %1, off offset:112" :: "v"(lds), "v"(ga) : "memory");
}

// --------------------------- conversion kernels ----------------------------
__global__ void k_cvt(const float* in, unsigned short* out, size_t n) {
  for (size_t i = blockIdx.x * (size_t)blockDim.x + threadIdx.x; i < n;
       i += (size_t)gridDim.x * blockDim.x)
    out[i] = f2bf(in[i]);
}

// in[mat][r][c] (f32) -> out[mat][c][r] (bf16)
__global__ void k_cvt_t(const float* in, unsigned short* out, int R, int C, int nmat) {
  size_t total = (size_t)nmat * R * C;
  for (size_t i = blockIdx.x * (size_t)blockDim.x + threadIdx.x; i < total;
       i += (size_t)gridDim.x * blockDim.x) {
    size_t mat = i / ((size_t)R * C);
    size_t rem = i - mat * (size_t)R * C;
    int r = (int)(rem / C), c = (int)(rem % C);
    out[mat * (size_t)R * C + (size_t)c * R + r] = f2bf(in[i]);
  }
}

// 8 WMMAs of one phase: two A rows against four B tiles.
#define PHASE8(A0, A1, Bt0, Bt1, Bt2, Bt3)                      \
  do {                                                          \
    acc0[0] = wmma_bf16(A0, Bt0, acc0[0]);                      \
    acc1[0] = wmma_bf16(A1, Bt0, acc1[0]);                      \
    acc0[1] = wmma_bf16(A0, Bt1, acc0[1]);                      \
    acc1[1] = wmma_bf16(A1, Bt1, acc1[1]);                      \
    acc0[2] = wmma_bf16(A0, Bt2, acc0[2]);                      \
    acc1[2] = wmma_bf16(A1, Bt2, acc1[2]);                      \
    acc0[3] = wmma_bf16(A0, Bt3, acc0[3]);                      \
    acc1[3] = wmma_bf16(A1, Bt3, acc1[3]);                      \
  } while (0)

// ------------------------------ QKV GEMM -----------------------------------
// grid: (B*L/32, H, 3). One wave computes a 32x64 tile; ping/pong k-loop.
__global__ void __launch_bounds__(32, 1)
k_qkv(const unsigned short* __restrict__ xb,
      const unsigned short* __restrict__ Wqt,
      const unsigned short* __restrict__ Wkt,
      const unsigned short* __restrict__ Wvt,
      const float* __restrict__ bq, const float* __restrict__ bk,
      const float* __restrict__ bv,
      unsigned short* __restrict__ Qb,
      unsigned short* __restrict__ Kb,
      unsigned short* __restrict__ Vt) {
  int rt = blockIdx.x, h = blockIdx.y, z = blockIdx.z;
  int lane = threadIdx.x & 31;
  int row0 = rt * 32;
  int b  = row0 / L_;
  int l0 = row0 - b * L_;
  const unsigned short* Wt = (z == 0 ? Wqt : z == 1 ? Wkt : Wvt) + (size_t)h * HD_ * D_;
  const float* bias = (z == 0 ? bq : z == 1 ? bk : bv) + h * HD_;

  v8f acc0[4], acc1[4];
#pragma unroll
  for (int t = 0; t < 4; ++t) { acc0[t] = vzero(); acc1[t] = vzero(); }

  // phase-A operands for k=0
  v16bf a0A = load_A_g(xb, D_, row0,      0);
  v16bf a1A = load_A_g(xb, D_, row0 + 16, 0);
  v16bf bA0 = load_B_g(Wt, D_,  0, 0);
  v16bf bA1 = load_B_g(Wt, D_, 16, 0);
  v16bf bA2 = load_B_g(Wt, D_, 32, 0);
  v16bf bA3 = load_B_g(Wt, D_, 48, 0);

  for (int k0 = 0; k0 < D_; k0 += 64) {
    // issue phase-B loads (k0+32) while phase-A computes
    v16bf a0B = load_A_g(xb, D_, row0,      k0 + 32);
    v16bf a1B = load_A_g(xb, D_, row0 + 16, k0 + 32);
    v16bf bB0 = load_B_g(Wt, D_,  0, k0 + 32);
    v16bf bB1 = load_B_g(Wt, D_, 16, k0 + 32);
    v16bf bB2 = load_B_g(Wt, D_, 32, k0 + 32);
    v16bf bB3 = load_B_g(Wt, D_, 48, k0 + 32);
    PHASE8(a0A, a1A, bA0, bA1, bA2, bA3);
    if (k0 + 64 < D_) {                     // issue next phase-A loads
      a0A = load_A_g(xb, D_, row0,      k0 + 64);
      a1A = load_A_g(xb, D_, row0 + 16, k0 + 64);
      bA0 = load_B_g(Wt, D_,  0, k0 + 64);
      bA1 = load_B_g(Wt, D_, 16, k0 + 64);
      bA2 = load_B_g(Wt, D_, 32, k0 + 64);
      bA3 = load_B_g(Wt, D_, 48, k0 + 64);
    }
    PHASE8(a0B, a1B, bB0, bB1, bB2, bB3);
  }

  int n = lane & 15, mb = (lane >> 4) << 3;
  size_t bh = (size_t)(b * H_ + h);
#pragma unroll
  for (int s = 0; s < 2; ++s) {
    v8f* acc = s ? acc1 : acc0;
    int ls = l0 + s * 16;
#pragma unroll
    for (int t = 0; t < 4; ++t) {
      float bvv = bias[t * 16 + n];
      if (z == 2) {                          // V stored transposed [b][h][d][l]
        int d = t * 16 + n;
        unsigned short* dst = Vt + bh * HD_ * L_ + (size_t)d * L_ + ls + mb;
        u32x4 pk;
#pragma unroll
        for (int i = 0; i < 4; ++i) {
          unsigned int lo = f2bf(acc[t][2 * i]     + bvv);
          unsigned int hi = f2bf(acc[t][2 * i + 1] + bvv);
          pk[i] = lo | (hi << 16);
        }
        *(u32x4*)dst = pk;
      } else {                               // Q,K stored [b][h][l][d]
        unsigned short* outp = (z == 0 ? Qb : Kb) + bh * (size_t)L_ * HD_;
#pragma unroll
        for (int r = 0; r < 8; ++r)
          outp[(size_t)(ls + mb + r) * HD_ + t * 16 + n] = f2bf(acc[t][r] + bvv);
      }
    }
  }
}

// --------------------------- flash attention -------------------------------
// grid: (L/16, B*H). One wave owns a 16-query tile, streams 32-key tiles.
// K tiles are staged into LDS with async loads, double buffered.
__global__ void __launch_bounds__(32, 1)
k_attn(const unsigned short* __restrict__ Qb,
       const unsigned short* __restrict__ Kb,
       const unsigned short* __restrict__ Vt,
       const unsigned char* __restrict__ pad,
       const int* __restrict__ causal_p,
       unsigned short* __restrict__ Zb) {
  int qt = blockIdx.x, bh = blockIdx.y;
  int b = bh >> 4, h = bh & 15;              // H_ == 16
  int lane = threadIdx.x & 31;
  int q0 = qt * 16;
  const unsigned short* Qh = Qb + (size_t)bh * L_ * HD_;
  const unsigned short* Kh = Kb + (size_t)bh * L_ * HD_;
  const unsigned short* Vh = Vt + (size_t)bh * HD_ * L_;
  int causal = causal_p[0];
  const float scale = 0.125f;                // 1/sqrt(64)

  v16bf qa0 = load_A_g(Qh, HD_, q0, 0);
  v16bf qa1 = load_A_g(Qh, HD_, q0, 32);

  v8f o[4];
#pragma unroll
  for (int t = 0; t < 4; ++t) o[t] = vzero();
  float mrow[8], lrow[8];
#pragma unroll
  for (int r = 0; r < 8; ++r) { mrow[r] = -1e30f; lrow[r] = 0.f; }

  int n = lane & 15, mb = (lane >> 4) << 3;
  __shared__ __align__(16) unsigned short kbuf[2][32 * HD_];  // 2 x 4KB K tiles
  __shared__ __align__(32) unsigned short pl[16 * 32];        // P bounce buffer
  int kend = causal ? (q0 + 16) : L_;

  // prime tile 0 into buffer 0 (lane l copies K row j0+l, 128B)
  async_copy_row128(Kh + (size_t)lane * HD_, &kbuf[0][lane * HD_]);

  for (int j0 = 0; j0 < kend; j0 += 32) {
    int cb = (j0 >> 5) & 1;
    if (j0 + 32 < kend) {                    // issue next tile, then wait <=8
      async_copy_row128(Kh + (size_t)(j0 + 32 + lane) * HD_,
                        &kbuf[cb ^ 1][lane * HD_]);
      __builtin_prefetch(Vh + j0 + 32, 0, 1);
      asm volatile("s_wait_asynccnt 8" ::: "memory");
    } else {
      asm volatile("s_wait_asynccnt 0" ::: "memory");
    }

    // ---- S = Q K^T (16x32 tile, depth 64 -> 4 WMMAs, K from LDS) ----
    v8f s0 = vzero(), s1 = vzero();
    {
      const unsigned short* kt = &kbuf[cb][0];
      v16bf kb0 = load_B_g(kt, HD_, 0, 0);
      v16bf kb1 = load_B_g(kt, HD_, 0, 32);
      s0 = wmma_bf16(qa0, kb0, s0);
      s0 = wmma_bf16(qa1, kb1, s0);
      v16bf kb2 = load_B_g(kt, HD_, 16, 0);
      v16bf kb3 = load_B_g(kt, HD_, 16, 32);
      s1 = wmma_bf16(qa0, kb2, s1);
      s1 = wmma_bf16(qa1, kb3, s1);
    }

    // ---- mask + online softmax (C layout: row m = mb+r, col n) ----
    int key0 = j0 + n, key1 = j0 + 16 + n;
    bool pm0 = pad[(size_t)b * L_ + key0] != 0;
    bool pm1 = pad[(size_t)b * L_ + key1] != 0;
    float p0[8], p1[8], rmax[8], rsum[8];
#pragma unroll
    for (int r = 0; r < 8; ++r) {
      int qi = q0 + mb + r;
      float v0 = s0[r] * scale, v1 = s1[r] * scale;
      bool ok0 = pm0 && (!causal || key0 <= qi);
      bool ok1 = pm1 && (!causal || key1 <= qi);
      p0[r] = ok0 ? v0 : -1e30f;
      p1[r] = ok1 ? v1 : -1e30f;
      rmax[r] = fmaxf(p0[r], p1[r]);
    }
#pragma unroll
    for (int off = 1; off < 16; off <<= 1)
#pragma unroll
      for (int r = 0; r < 8; ++r)
        rmax[r] = fmaxf(rmax[r], __shfl_xor(rmax[r], off, 32));

    float alpha[8];
#pragma unroll
    for (int r = 0; r < 8; ++r) {
      float mnew = fmaxf(mrow[r], rmax[r]);
      alpha[r] = __expf(mrow[r] - mnew);
      mrow[r] = mnew;
      float e0 = __expf(p0[r] - mnew);
      float e1 = __expf(p1[r] - mnew);
      rsum[r] = e0 + e1;
      pl[(mb + r) * 32 + n]      = f2bf(e0);   // C-layout -> LDS row-major
      pl[(mb + r) * 32 + 16 + n] = f2bf(e1);
    }
#pragma unroll
    for (int off = 1; off < 16; off <<= 1)
#pragma unroll
      for (int r = 0; r < 8; ++r)
        rsum[r] += __shfl_xor(rsum[r], off, 32);
#pragma unroll
    for (int r = 0; r < 8; ++r)
      lrow[r] = lrow[r] * alpha[r] + rsum[r];
#pragma unroll
    for (int t = 0; t < 4; ++t)
#pragma unroll
      for (int r = 0; r < 8; ++r)
        o[t][r] *= alpha[r];

    asm volatile("s_wait_dscnt 0" ::: "memory");   // DS in-order per wave; belt & braces

    // ---- reload P as A-operand from LDS ----
    v16bf pa;
    {
      int m = lane & 15, kb = (lane >> 4) << 3;
      const unsigned short* pp = &pl[m * 32 + kb];
      OpU r2;
      r2.q[0] = *(const u32x4*)pp;
      r2.q[1] = *(const u32x4*)(pp + 16);
      pa = r2.v;
    }
    // ---- O += P V  (V transposed: rows = d, cols = keys) ----
#pragma unroll
    for (int t = 0; t < 4; ++t) {
      v16bf vb = load_B_g(Vh, L_, t * 16, j0);
      o[t] = wmma_bf16(pa, vb, o[t]);
    }
  }

  // ---- normalize and store Z [b][l][h*64+d] bf16 ----
  float inv[8];
#pragma unroll
  for (int r = 0; r < 8; ++r) inv[r] = 1.0f / lrow[r];
#pragma unroll
  for (int t = 0; t < 4; ++t)
#pragma unroll
    for (int r = 0; r < 8; ++r) {
      size_t row = (size_t)b * L_ + q0 + mb + r;
      Zb[row * D_ + h * HD_ + t * 16 + n] = f2bf(o[t][r] * inv[r]);
    }
}

// --------------------------- output projection -----------------------------
// grid: (B*L/32, D/64). out = Z @ Wo, f32 result, 32x64 tile per wave.
__global__ void __launch_bounds__(32, 1)
k_proj(const unsigned short* __restrict__ Zb,
       const unsigned short* __restrict__ Wot,
       float* __restrict__ out) {
  int rt = blockIdx.x, ct = blockIdx.y;
  int lane = threadIdx.x & 31;
  int row0 = rt * 32;
  v8f acc0[4], acc1[4];
#pragma unroll
  for (int t = 0; t < 4; ++t) { acc0[t] = vzero(); acc1[t] = vzero(); }

  const unsigned short* Wc = Wot + (size_t)ct * 64 * D_;
  v16bf a0A = load_A_g(Zb, D_, row0,      0);
  v16bf a1A = load_A_g(Zb, D_, row0 + 16, 0);
  v16bf bA0 = load_B_g(Wc, D_,  0, 0);
  v16bf bA1 = load_B_g(Wc, D_, 16, 0);
  v16bf bA2 = load_B_g(Wc, D_, 32, 0);
  v16bf bA3 = load_B_g(Wc, D_, 48, 0);

  for (int k0 = 0; k0 < D_; k0 += 64) {
    v16bf a0B = load_A_g(Zb, D_, row0,      k0 + 32);
    v16bf a1B = load_A_g(Zb, D_, row0 + 16, k0 + 32);
    v16bf bB0 = load_B_g(Wc, D_,  0, k0 + 32);
    v16bf bB1 = load_B_g(Wc, D_, 16, k0 + 32);
    v16bf bB2 = load_B_g(Wc, D_, 32, k0 + 32);
    v16bf bB3 = load_B_g(Wc, D_, 48, k0 + 32);
    PHASE8(a0A, a1A, bA0, bA1, bA2, bA3);
    if (k0 + 64 < D_) {
      a0A = load_A_g(Zb, D_, row0,      k0 + 64);
      a1A = load_A_g(Zb, D_, row0 + 16, k0 + 64);
      bA0 = load_B_g(Wc, D_,  0, k0 + 64);
      bA1 = load_B_g(Wc, D_, 16, k0 + 64);
      bA2 = load_B_g(Wc, D_, 32, k0 + 64);
      bA3 = load_B_g(Wc, D_, 48, k0 + 64);
    }
    PHASE8(a0B, a1B, bB0, bB1, bB2, bB3);
  }

  int n = lane & 15, mb = (lane >> 4) << 3;
#pragma unroll
  for (int s = 0; s < 2; ++s) {
    v8f* acc = s ? acc1 : acc0;
#pragma unroll
    for (int t = 0; t < 4; ++t)
#pragma unroll
      for (int r = 0; r < 8; ++r)
        out[(size_t)(row0 + s * 16 + mb + r) * D_ + ct * 64 + t * 16 + n] = acc[t][r];
  }
}

// ------------------------------- launcher ----------------------------------
extern "C" void kernel_launch(void* const* d_in, const int* in_sizes, int n_in,
                              void* d_out, int out_size, void* d_ws, size_t ws_size,
                              hipStream_t stream) {
  (void)in_sizes; (void)n_in; (void)out_size; (void)ws_size;
  const float* x  = (const float*)d_in[0];
  const float* Wq = (const float*)d_in[1];
  const float* bq = (const float*)d_in[2];
  const float* Wk = (const float*)d_in[3];
  const float* bk = (const float*)d_in[4];
  const float* Wv = (const float*)d_in[5];
  const float* bv = (const float*)d_in[6];
  const float* Wo = (const float*)d_in[7];
  const unsigned char* pad = (const unsigned char*)d_in[8];
  const int* causal = (const int*)d_in[9];
  float* out = (float*)d_out;

  char* ws = (char*)d_ws;
  size_t off = 0;
  auto take = [&](size_t bytes) -> char* {
    char* p = ws + off;
    off = (off + bytes + 255) & ~(size_t)255;
    return p;
  };
  unsigned short* xb  = (unsigned short*)take((size_t)B_ * L_ * D_ * 2);
  unsigned short* Wqt = (unsigned short*)take((size_t)H_ * HD_ * D_ * 2);
  unsigned short* Wkt = (unsigned short*)take((size_t)H_ * HD_ * D_ * 2);
  unsigned short* Wvt = (unsigned short*)take((size_t)H_ * HD_ * D_ * 2);
  unsigned short* Wot = (unsigned short*)take((size_t)D_ * D_ * 2);
  unsigned short* Qb  = (unsigned short*)take((size_t)B_ * H_ * L_ * HD_ * 2);
  unsigned short* Kb  = (unsigned short*)take((size_t)B_ * H_ * L_ * HD_ * 2);
  unsigned short* Vtb = (unsigned short*)take((size_t)B_ * H_ * HD_ * L_ * 2);
  unsigned short* Zb  = (unsigned short*)take((size_t)B_ * L_ * D_ * 2);

  k_cvt  <<<dim3(2048), dim3(256), 0, stream>>>(x, xb, (size_t)B_ * L_ * D_);
  k_cvt_t<<<dim3(1024), dim3(256), 0, stream>>>(Wq, Wqt, D_, HD_, H_);
  k_cvt_t<<<dim3(1024), dim3(256), 0, stream>>>(Wk, Wkt, D_, HD_, H_);
  k_cvt_t<<<dim3(1024), dim3(256), 0, stream>>>(Wv, Wvt, D_, HD_, H_);
  k_cvt_t<<<dim3(1024), dim3(256), 0, stream>>>(Wo, Wot, D_, D_, 1);

  k_qkv <<<dim3(B_ * L_ / 32, H_, 3), dim3(32), 0, stream>>>(
      xb, Wqt, Wkt, Wvt, bq, bk, bv, Qb, Kb, Vtb);
  k_attn<<<dim3(L_ / 16, B_ * H_), dim3(32), 0, stream>>>(
      Qb, Kb, Vtb, pad, causal, Zb);
  k_proj<<<dim3(B_ * L_ / 32, D_ / 64), dim3(32), 0, stream>>>(Zb, Wot, out);
}
